// GeometricGNN_24859270709373
// MI455X (gfx1250) — matrix operand order repaired and checked by reading, not compiled, not run on hardware
//
#include <hip/hip_runtime.h>
#include <hip/hip_bf16.h>

#define R_ 2048
#define A_ 16384
#define H_ 512
#define HEADS_ 4
#define HD_ 128
#define SD_ 28
#define KP_ 544   // padded K for output projection (540 -> 544)

typedef __attribute__((ext_vector_type(16))) __bf16 v16bf;
typedef __attribute__((ext_vector_type(8)))  float  v8f;

__device__ __forceinline__ v8f wmma_bf16(v16bf a, v16bf b, v8f c) {
  return __builtin_amdgcn_wmma_f32_16x16x32_bf16(false, a, false, b, (short)0, c, false, false);
}

// load 8 consecutive bf16 (16B aligned) into frag slots [base..base+7]
__device__ __forceinline__ void ld8bf(v16bf& f, int base, const __bf16* p) {
  union { uint4 v; __bf16 h[8]; } t;
  t.v = *reinterpret_cast<const uint4*>(p);
#pragma unroll
  for (int i = 0; i < 8; ++i) f[base + i] = t.h[i];
}

__device__ __forceinline__ float redmax16(float t) {
  t = fmaxf(t, __shfl_xor(t, 1, 32));
  t = fmaxf(t, __shfl_xor(t, 2, 32));
  t = fmaxf(t, __shfl_xor(t, 4, 32));
  t = fmaxf(t, __shfl_xor(t, 8, 32));
  return t;
}
__device__ __forceinline__ float redsum16(float t) {
  t += __shfl_xor(t, 1, 32);
  t += __shfl_xor(t, 2, 32);
  t += __shfl_xor(t, 4, 32);
  t += __shfl_xor(t, 8, 32);
  return t;
}

// ---------------- K0: init per-residue scratch ----------------
__global__ void k_init(int* ca_idx, int* cb_idx, int* cnt, int* has) {
  int r = blockIdx.x * 256 + threadIdx.x;
  if (r >= R_) return;
  ca_idx[r] = -1; cb_idx[r] = -1; cnt[r] = 0; has[r] = 0;
}

// ---------------- K1: atom scan (scatter) ----------------
__global__ void k_atoms(const int* __restrict__ type, const int* __restrict__ a2r,
                        int* ca_idx, int* cb_idx, int* cnt, int* has) {
  int a = blockIdx.x * 256 + threadIdx.x;
  if (a >= A_) return;
  int r = a2r[a];
  if (r < 0 || r >= R_) return;
  int t = type[a];
  atomicAdd(&cnt[r], 1);
  if (t < 3) atomicOr(&has[r], 1 << t);
  if (t == 1) atomicExch(&ca_idx[r], a);
  if (t == 4) atomicExch(&cb_idx[r], a);
}

__device__ __forceinline__ void cross3(const float* a, const float* b, float* c) {
  c[0] = a[1] * b[2] - a[2] * b[1];
  c[1] = a[2] * b[0] - a[0] * b[2];
  c[2] = a[0] * b[1] - a[1] * b[0];
}

// ---------------- K2: residue positions / mask / frames ----------------
__global__ void k_res(const float* __restrict__ pos, const int* ca_idx, const int* cb_idx,
                      const int* cnt, const int* has, int* mask,
                      float* pCA, float* pCB, float* fr) {
  int r = blockIdx.x * 256 + threadIdx.x;
  if (r >= R_) return;
  int ci = ca_idx[r], bi = cb_idx[r];
  float ca[3] = {0.f, 0.f, 0.f}, cb[3] = {0.f, 0.f, 0.f};
  if (ci >= 0) { ca[0] = pos[ci * 3]; ca[1] = pos[ci * 3 + 1]; ca[2] = pos[ci * 3 + 2]; }
  if (bi >= 0) { cb[0] = pos[bi * 3]; cb[1] = pos[bi * 3 + 1]; cb[2] = pos[bi * 3 + 2]; }
  if (fabsf(cb[0]) + fabsf(cb[1]) + fabsf(cb[2]) < 1e-6f) { cb[0] = ca[0]; cb[1] = ca[1]; cb[2] = ca[2]; }
  mask[r] = (cnt[r] >= 3 && has[r] == 7) ? 1 : 0;
#pragma unroll
  for (int d = 0; d < 3; ++d) { pCA[r * 3 + d] = ca[d]; pCB[r * 3 + d] = cb[d]; }
  float e1[3] = {cb[0] - ca[0], cb[1] - ca[1], cb[2] - ca[2]};
  float n1 = sqrtf(e1[0] * e1[0] + e1[1] * e1[1] + e1[2] * e1[2]);
  float e1u[3];
  if (n1 > 1e-6f) { float inv = 1.f / fmaxf(n1, 1e-6f); e1u[0] = e1[0] * inv; e1u[1] = e1[1] * inv; e1u[2] = e1[2] * inv; }
  else { e1u[0] = e1[0]; e1u[1] = e1[1]; e1u[2] = e1[2]; }
  float zax[3] = {0.f, 0.f, 1.f}, yax[3] = {0.f, 1.f, 0.f};
  float e2a[3], e2b[3];
  cross3(e1u, zax, e2a);
  cross3(e1u, yax, e2b);
  float n2a = sqrtf(e2a[0] * e2a[0] + e2a[1] * e2a[1] + e2a[2] * e2a[2]);
  float n2b = sqrtf(e2b[0] * e2b[0] + e2b[1] * e2b[1] + e2b[2] * e2b[2]);
  bool useb = n2a < 1e-6f;
  float e2[3] = {useb ? e2b[0] : e2a[0], useb ? e2b[1] : e2a[1], useb ? e2b[2] : e2a[2]};
  float n2 = useb ? n2b : n2a;
  float invn2 = 1.f / fmaxf(n2, 1e-6f);
  float e2u[3] = {e2[0] * invn2, e2[1] * invn2, e2[2] * invn2};
  float e3[3];
  cross3(e1u, e2u, e3);
  bool ok = (r < R_ - 1) && (n1 > 1e-6f) && (n2 > 1e-6f);
#pragma unroll
  for (int i = 0; i < 3; ++i) {
    fr[r * 9 + i * 3 + 0] = ok ? e1u[i] : (i == 0 ? 1.f : 0.f);
    fr[r * 9 + i * 3 + 1] = ok ? e2u[i] : (i == 1 ? 1.f : 0.f);
    fr[r * 9 + i * 3 + 2] = ok ? e3[i]  : (i == 2 ? 1.f : 0.f);
  }
}

// ---------------- K2b: transpose + bf16-convert weights ----------------
// z=0..2: WT[n*512+k] = W[k*512+n] for Wq/Wk/Wv. z=3: WoT[n*544+k] (K padded w/ 0)
__global__ void k_prepw(const float* __restrict__ Wq, const float* __restrict__ Wk,
                        const float* __restrict__ Wv, const float* __restrict__ Wo,
                        __bf16* WqT, __bf16* WkT, __bf16* WvT, __bf16* WoT) {
  int z = blockIdx.y;
  int idx = blockIdx.x * 256 + threadIdx.x;
  if (z < 3) {
    if (idx >= H_ * H_) return;
    int n = idx >> 9, k = idx & (H_ - 1);
    const float* W = (z == 0) ? Wq : ((z == 1) ? Wk : Wv);
    __bf16* T = (z == 0) ? WqT : ((z == 1) ? WkT : WvT);
    T[(size_t)n * H_ + k] = (__bf16)W[(size_t)k * H_ + n];
  } else {
    if (idx >= H_ * KP_) return;
    int n = idx / KP_, k = idx % KP_;
    WoT[(size_t)n * KP_ + k] = (k < H_ + SD_) ? (__bf16)Wo[(size_t)k * H_ + n] : (__bf16)0.f;
  }
}

// ---------------- K3: QKV projection (bf16 WMMA) ----------------
// grid (128 m-tiles, 8 n-strips of 64, 3 matrices), block 32 (one wave)
// q/k stored row-major [R_][H_]; v stored transposed vT[H_][R_]
__global__ void __launch_bounds__(32)
k_qkv(const float* __restrict__ X, const int* __restrict__ ca_idx,
      const __bf16* __restrict__ WqT, const float* __restrict__ bq,
      const __bf16* __restrict__ WkT, const float* __restrict__ bk,
      const __bf16* __restrict__ WvT, const float* __restrict__ bv,
      __bf16* qb, __bf16* kb, __bf16* vT) {
  int lane = threadIdx.x, half = lane >> 4, Nl = lane & 15;
  int l0 = blockIdx.x * 16, n0 = blockIdx.y * 64;
  const __bf16* WT; const float* bias;
  if (blockIdx.z == 0)      { WT = WqT; bias = bq; }
  else if (blockIdx.z == 1) { WT = WkT; bias = bk; }
  else                      { WT = WvT; bias = bv; }
  int ci = ca_idx[l0 + Nl];
  const float* xrow = (ci >= 0) ? (X + (size_t)ci * H_) : nullptr;
  v8f acc[4];
#pragma unroll
  for (int t = 0; t < 4; ++t)
#pragma unroll
    for (int j = 0; j < 8; ++j) acc[t][j] = 0.f;

  for (int k0 = 0; k0 < H_; k0 += 32) {
    v16bf Af;
    if (xrow) {
      const float* p0 = xrow + k0 + half * 8;
      const float* p1 = p0 + 16;
      float4 f0 = *(const float4*)p0, f1 = *(const float4*)(p0 + 4);
      float4 f2 = *(const float4*)p1, f3 = *(const float4*)(p1 + 4);
      Af[0] = (__bf16)f0.x; Af[1] = (__bf16)f0.y; Af[2] = (__bf16)f0.z; Af[3] = (__bf16)f0.w;
      Af[4] = (__bf16)f1.x; Af[5] = (__bf16)f1.y; Af[6] = (__bf16)f1.z; Af[7] = (__bf16)f1.w;
      Af[8] = (__bf16)f2.x; Af[9] = (__bf16)f2.y; Af[10] = (__bf16)f2.z; Af[11] = (__bf16)f2.w;
      Af[12] = (__bf16)f3.x; Af[13] = (__bf16)f3.y; Af[14] = (__bf16)f3.z; Af[15] = (__bf16)f3.w;
    } else {
#pragma unroll
      for (int e = 0; e < 16; ++e) Af[e] = (__bf16)0.f;
    }
#pragma unroll
    for (int nt = 0; nt < 4; ++nt) {
      int col = n0 + nt * 16 + Nl;
      const __bf16* wp = WT + (size_t)col * H_ + k0 + half * 8;
      v16bf Bf;
      ld8bf(Bf, 0, wp); ld8bf(Bf, 8, wp + 16);
      acc[nt] = wmma_bf16(Af, Bf, acc[nt]);
    }
  }
  if (blockIdx.z < 2) {
    __bf16* out = (blockIdx.z == 0) ? qb : kb;
#pragma unroll
    for (int nt = 0; nt < 4; ++nt) {
      int col = n0 + nt * 16 + Nl;
      float bv_ = bias[col];
#pragma unroll
      for (int j = 0; j < 8; ++j)
        out[(size_t)(l0 + j + 8 * half) * H_ + col] = (__bf16)(acc[nt][j] + bv_);
    }
  } else {
    // transposed store: rows l0+8*half..+7 are contiguous in vT -> one b128/lane/tile
#pragma unroll
    for (int nt = 0; nt < 4; ++nt) {
      int col = n0 + nt * 16 + Nl;
      float bv_ = bias[col];
      union { uint4 u; __bf16 h[8]; } t;
#pragma unroll
      for (int j = 0; j < 8; ++j) t.h[j] = (__bf16)(acc[nt][j] + bv_);
      *reinterpret_cast<uint4*>(vT + (size_t)col * R_ + l0 + 8 * half) = t.u;
    }
  }
}

// ---------------- K4: flash attention per (l-tile, head) ----------------
// grid (128, 4), block 32 (one wave). Online softmax; bias = cb - alpha@ca fused.
// feat_node written into padded feat_all buffer fa[R_][KP_] cols [0,512)
__global__ void __launch_bounds__(32)
k_attn(const __bf16* __restrict__ qb, const __bf16* __restrict__ kb,
       const __bf16* __restrict__ vT, const int* __restrict__ mask,
       const float* __restrict__ pCA, const float* __restrict__ pCB,
       const float* __restrict__ fr, __bf16* fa, float* spatial) {
  __shared__ float ptile[16 * 32];
  int lane = threadIdx.x, half = lane >> 4, Nl = lane & 15;
  int l0 = blockIdx.x * 16, h = blockIdx.y;
  const __bf16* qrow = qb + (size_t)(l0 + Nl) * H_ + h * HD_;

  float mst[8], lst[8], racc[8][3];
  v8f O[8];
#pragma unroll
  for (int j = 0; j < 8; ++j) {
    mst[j] = -3e38f; lst[j] = 0.f;
    racc[j][0] = racc[j][1] = racc[j][2] = 0.f;
  }
#pragma unroll
  for (int nt = 0; nt < 8; ++nt)
#pragma unroll
    for (int j = 0; j < 8; ++j) O[nt][j] = 0.f;

  for (int kc = 0; kc < R_; kc += 32) {
    // ---- logits for two 16-wide column sub-tiles ----
    v8f a0, a1;
#pragma unroll
    for (int j = 0; j < 8; ++j) { a0[j] = 0.f; a1[j] = 0.f; }
#pragma unroll
    for (int kd = 0; kd < HD_; kd += 32) {
      v16bf Af;
      ld8bf(Af, 0, qrow + kd + half * 8);
      ld8bf(Af, 8, qrow + kd + 16 + half * 8);
      v16bf Bf;
      const __bf16* kp = kb + (size_t)(kc + Nl) * H_ + h * HD_ + kd + half * 8;
      ld8bf(Bf, 0, kp); ld8bf(Bf, 8, kp + 16);
      a0 = wmma_bf16(Af, Bf, a0);
      const __bf16* kp2 = kp + (size_t)16 * H_;
      ld8bf(Bf, 0, kp2); ld8bf(Bf, 8, kp2 + 16);
      a1 = wmma_bf16(Af, Bf, a1);
    }
    int c0 = kc + Nl, c1 = kc + 16 + Nl;
    bool ok0 = mask[c0] != 0, ok1 = mask[c1] != 0;
    float ca0[3], ca1[3];
#pragma unroll
    for (int d = 0; d < 3; ++d) { ca0[d] = pCA[c0 * 3 + d]; ca1[d] = pCA[c1 * 3 + d]; }

#pragma unroll
    for (int j = 0; j < 8; ++j) {
      float x0 = ok0 ? a0[j] : -3e38f;
      float x1 = ok1 ? a1[j] : -3e38f;
      float tmax = redmax16(fmaxf(x0, x1));
      float mnew = fmaxf(mst[j], tmax);
      float corr = __expf(mst[j] - mnew);
      mst[j] = mnew;
      float p0 = ok0 ? __expf(a0[j] - mnew) : 0.f;
      float p1 = ok1 ? __expf(a1[j] - mnew) : 0.f;
      int row = j + 8 * half;
      ptile[row * 32 + Nl] = p0;
      ptile[row * 32 + 16 + Nl] = p1;
      float ts = redsum16(p0 + p1);
      lst[j] = lst[j] * corr + ts;
#pragma unroll
      for (int d = 0; d < 3; ++d)
        racc[j][d] = racc[j][d] * corr + p0 * ca0[d] + p1 * ca1[d];
#pragma unroll
      for (int nt = 0; nt < 8; ++nt) O[nt][j] *= corr;
    }
    __syncthreads();
    // ---- O += P @ V (V transposed: both operand loads are b128) ----
    v16bf Ap;
    {
      const float* pr = &ptile[Nl * 32 + half * 8];
#pragma unroll
      for (int i = 0; i < 8; ++i) {
        Ap[i] = (__bf16)pr[i];
        Ap[8 + i] = (__bf16)pr[16 + i];
      }
    }
#pragma unroll
    for (int nt = 0; nt < 8; ++nt) {
      v16bf Bf;
      const __bf16* vp = vT + (size_t)(h * HD_ + nt * 16 + Nl) * R_ + kc + half * 8;
      ld8bf(Bf, 0, vp); ld8bf(Bf, 8, vp + 16);
      O[nt] = wmma_bf16(Ap, Bf, O[nt]);
    }
    __syncthreads();
  }
  // ---- epilogue ----
  float inv[8];
#pragma unroll
  for (int j = 0; j < 8; ++j) inv[j] = (lst[j] > 0.f) ? (1.f / lst[j]) : 0.f;
#pragma unroll
  for (int nt = 0; nt < 8; ++nt)
#pragma unroll
    for (int j = 0; j < 8; ++j)
      fa[(size_t)(l0 + j + 8 * half) * KP_ + h * HD_ + nt * 16 + Nl] =
          (__bf16)(O[nt][j] * inv[j]);
#pragma unroll
  for (int j = 0; j < 8; ++j)
#pragma unroll
    for (int d = 0; d < 3; ++d) racc[j][d] = redsum16(racc[j][d]);

  if (Nl == 0) {
#pragma unroll
    for (int j = 0; j < 8; ++j) {
      int r = l0 + j + 8 * half;
      float b3[3];
#pragma unroll
      for (int d = 0; d < 3; ++d) b3[d] = pCB[r * 3 + d] - racc[j][d] * inv[j];
      float pts[3];
#pragma unroll
      for (int i = 0; i < 3; ++i)
        pts[i] = fr[r * 9 + i * 3 + 0] * b3[0] + fr[r * 9 + i * 3 + 1] * b3[1] +
                 fr[r * 9 + i * 3 + 2] * b3[2];
      float dist = sqrtf(b3[0] * b3[0] + b3[1] * b3[1] + b3[2] * b3[2]);
      float pn = sqrtf(pts[0] * pts[0] + pts[1] * pts[1] + pts[2] * pts[2]) + 1e-10f;
#pragma unroll
      for (int i = 0; i < 3; ++i) {
        spatial[r * SD_ + h * 3 + i] = pts[i];
        spatial[r * SD_ + 16 + h * 3 + i] = pts[i] / pn;
      }
      spatial[r * SD_ + 12 + h] = dist;
    }
  }
}

// ---------------- K4b: fill spatial cols [512,544) of feat_all ----------------
__global__ void k_tail(const float* __restrict__ spatial, __bf16* fa) {
  int idx = blockIdx.x * 256 + threadIdx.x;
  if (idx >= R_ * 32) return;
  int r = idx >> 5, c = idx & 31;
  fa[(size_t)r * KP_ + H_ + c] = (c < SD_) ? (__bf16)spatial[r * SD_ + c] : (__bf16)0.f;
}

// ---------------- K5: output projection + ReLU + LN1 + mask + residual + LN2 ----------------
// grid 128 (l-tiles), block 128 (4 waves; wave w owns cols [w*128, w*128+128))
__global__ void __launch_bounds__(128)
k_out(const __bf16* __restrict__ fa, const __bf16* __restrict__ WoT,
      const float* __restrict__ bo,
      const float* __restrict__ g1, const float* __restrict__ b1,
      const float* __restrict__ g2, const float* __restrict__ b2,
      const int* __restrict__ mask, const int* __restrict__ ca_idx,
      const float* __restrict__ X, float* __restrict__ out) {
  __shared__ float s1[16], s2[16];
  int tid = threadIdx.x, lane = tid & 31, w = tid >> 5;
  int half = lane >> 4, Nl = lane & 15;
  int l0 = blockIdx.x * 16;
  int arow = l0 + Nl;

  v8f acc[8];
#pragma unroll
  for (int nt = 0; nt < 8; ++nt)
#pragma unroll
    for (int j = 0; j < 8; ++j) acc[nt][j] = 0.f;

  for (int ks = 0; ks < 17; ++ks) {
    int k0 = ks * 32;
    v16bf Af;
    const __bf16* ap = fa + (size_t)arow * KP_ + k0 + half * 8;
    ld8bf(Af, 0, ap); ld8bf(Af, 8, ap + 16);
#pragma unroll
    for (int nt = 0; nt < 8; ++nt) {
      int col = (w * 8 + nt) * 16 + Nl;
      v16bf Bf;
      const __bf16* wp = WoT + (size_t)col * KP_ + k0 + half * 8;
      ld8bf(Bf, 0, wp); ld8bf(Bf, 8, wp + 16);
      acc[nt] = wmma_bf16(Af, Bf, acc[nt]);
    }
  }
  // bias + relu
  float hv[8][8];
#pragma unroll
  for (int nt = 0; nt < 8; ++nt) {
    int col = (w * 8 + nt) * 16 + Nl;
    float bb = bo[col];
#pragma unroll
    for (int j = 0; j < 8; ++j) hv[nt][j] = fmaxf(acc[nt][j] + bb, 0.f);
  }
  // LN1 row stats
  if (tid < 16) { s1[tid] = 0.f; s2[tid] = 0.f; }
  __syncthreads();
#pragma unroll
  for (int j = 0; j < 8; ++j) {
    float ps = 0.f, ps2 = 0.f;
#pragma unroll
    for (int nt = 0; nt < 8; ++nt) { ps += hv[nt][j]; ps2 += hv[nt][j] * hv[nt][j]; }
    atomicAdd(&s1[j + 8 * half], ps);
    atomicAdd(&s2[j + 8 * half], ps2);
  }
  __syncthreads();
#pragma unroll
  for (int j = 0; j < 8; ++j) {
    int r = j + 8 * half;
    float mu = s1[r] * (1.f / H_);
    float var = s2[r] * (1.f / H_) - mu * mu;
    float is = rsqrtf(var + 1e-5f);
    bool mk = mask[l0 + r] != 0;
#pragma unroll
    for (int nt = 0; nt < 8; ++nt) {
      int col = (w * 8 + nt) * 16 + Nl;
      hv[nt][j] = mk ? ((hv[nt][j] - mu) * is * g1[col] + b1[col]) : 0.f;
    }
  }
  __syncthreads();
  if (tid < 16) { s1[tid] = 0.f; s2[tid] = 0.f; }
  __syncthreads();
  // residual add + LN2 stats
#pragma unroll
  for (int j = 0; j < 8; ++j) {
    int rr = l0 + j + 8 * half;
    int ci = ca_idx[rr];
    float ps = 0.f, ps2 = 0.f;
#pragma unroll
    for (int nt = 0; nt < 8; ++nt) {
      int col = (w * 8 + nt) * 16 + Nl;
      float xv = (ci >= 0) ? X[(size_t)ci * H_ + col] : 0.f;
      float y = xv + hv[nt][j];
      hv[nt][j] = y;
      ps += y; ps2 += y * y;
    }
    atomicAdd(&s1[j + 8 * half], ps);
    atomicAdd(&s2[j + 8 * half], ps2);
  }
  __syncthreads();
#pragma unroll
  for (int j = 0; j < 8; ++j) {
    int r = j + 8 * half;
    float mu = s1[r] * (1.f / H_);
    float var = s2[r] * (1.f / H_) - mu * mu;
    float is = rsqrtf(var + 1e-5f);
#pragma unroll
    for (int nt = 0; nt < 8; ++nt) {
      int col = (w * 8 + nt) * 16 + Nl;
      out[(size_t)(l0 + r) * H_ + col] = (hv[nt][j] - mu) * is * g2[col] + b2[col];
    }
  }
}

// ---------------- host launcher ----------------
extern "C" void kernel_launch(void* const* d_in, const int* in_sizes, int n_in,
                              void* d_out, int out_size, void* d_ws, size_t ws_size,
                              hipStream_t stream) {
  const float* node_features  = (const float*)d_in[0];
  const float* node_positions = (const float*)d_in[1];
  const float* Wq = (const float*)d_in[2];  const float* bq = (const float*)d_in[3];
  const float* Wk = (const float*)d_in[4];  const float* bk = (const float*)d_in[5];
  const float* Wv = (const float*)d_in[6];  const float* bv = (const float*)d_in[7];
  const float* Wo = (const float*)d_in[8];  const float* bo = (const float*)d_in[9];
  const float* g1 = (const float*)d_in[10]; const float* b1 = (const float*)d_in[11];
  const float* g2 = (const float*)d_in[12]; const float* b2 = (const float*)d_in[13];
  const int* atom_type = (const int*)d_in[14];
  const int* a2r       = (const int*)d_in[15];

  char* ws = (char*)d_ws;
  // workspace layout (total ~10.6 MB)
  int*    ca_idx  = (int*)(ws + 0);
  int*    cb_idx  = (int*)(ws + 8192);
  int*    cnt     = (int*)(ws + 16384);
  int*    has     = (int*)(ws + 24576);
  int*    mask    = (int*)(ws + 32768);
  float*  pCA     = (float*)(ws + 40960);
  float*  pCB     = (float*)(ws + 65536);
  float*  frames  = (float*)(ws + 90112);
  float*  spatial = (float*)(ws + 163840);
  __bf16* WqT     = (__bf16*)(ws + 393216);    // 512KB
  __bf16* WkT     = (__bf16*)(ws + 917504);    // 512KB
  __bf16* WvT     = (__bf16*)(ws + 1441792);   // 512KB
  __bf16* WoT     = (__bf16*)(ws + 1966080);   // 544KB
  __bf16* qb      = (__bf16*)(ws + 2523136);   // 2MB
  __bf16* kb      = (__bf16*)(ws + 4620288);   // 2MB
  __bf16* vT      = (__bf16*)(ws + 6717440);   // 2MB (transposed [H_][R_])
  __bf16* fa      = (__bf16*)(ws + 8814592);   // 2.125MB (feat_all padded [R_][544])
  float*  out     = (float*)d_out;

  k_init<<<(R_ + 255) / 256, 256, 0, stream>>>(ca_idx, cb_idx, cnt, has);
  k_atoms<<<(A_ + 255) / 256, 256, 0, stream>>>(atom_type, a2r, ca_idx, cb_idx, cnt, has);
  k_res<<<(R_ + 255) / 256, 256, 0, stream>>>(node_positions, ca_idx, cb_idx, cnt, has,
                                              mask, pCA, pCB, frames);
  k_prepw<<<dim3((H_ * KP_ + 255) / 256, 4), 256, 0, stream>>>(Wq, Wk, Wv, Wo,
                                                               WqT, WkT, WvT, WoT);
  k_qkv<<<dim3(R_ / 16, H_ / 64, 3), 32, 0, stream>>>(node_features, ca_idx,
                                                      WqT, bq, WkT, bk, WvT, bv,
                                                      qb, kb, vT);
  k_attn<<<dim3(R_ / 16, HEADS_), 32, 0, stream>>>(qb, kb, vT, mask, pCA, pCB, frames,
                                                   fa, spatial);
  k_tail<<<(R_ * 32 + 255) / 256, 256, 0, stream>>>(spatial, fa);
  k_out<<<R_ / 16, 128, 0, stream>>>(fa, WoT, bo, g1, b1, g2, b2, mask, ca_idx,
                                     node_features, out);
}
